// SelfMultiheadAttnNormAdd_68796786147618
// MI455X (gfx1250) — matrix operand order repaired and checked by reading, not compiled, not run on hardware
//
#include <hip/hip_runtime.h>
#include <hip/hip_bf16.h>

// ---------------------------------------------------------------------------
// Fused self-attention block for MI455X (gfx1250, wave32, WMMA).
//  query[T,B,E] -> LN -> QKV proj -> MHA (flash softmax) -> out proj -> +query
//  T=2048 B=2 E=1024 H=16 HD=64
// Matmuls: v_wmma_f32_16x16x32_bf16.  Tile staging: global_load_async_to_lds
// (ASYNCcnt) for GEMM, Tensor Data Mover (TENSORcnt) for attention Q/K tiles.
// ---------------------------------------------------------------------------

typedef __attribute__((ext_vector_type(16))) __bf16 v16bf;
typedef __attribute__((ext_vector_type(8)))  __bf16 v8bf;
typedef __attribute__((ext_vector_type(8)))  float  v8f;
typedef __attribute__((ext_vector_type(4)))  unsigned int v4u;
typedef __attribute__((ext_vector_type(8)))  int v8i;
typedef __attribute__((ext_vector_type(4)))  int v4i;

#define DEV __device__ __forceinline__

DEV v8f wmma_bf16(v16bf a, v16bf b, v8f c) {
  // (neg_a, A, neg_b, B, c_mod, C, reuse_a, reuse_b)
  return __builtin_amdgcn_wmma_f32_16x16x32_bf16(false, a, false, b, (short)0, c,
                                                 false, false);
}

DEV v16bf cat16(v8bf lo, v8bf hi) {
  return __builtin_shufflevector(lo, hi, 0, 1, 2, 3, 4, 5, 6, 7,
                                 8, 9, 10, 11, 12, 13, 14, 15);
}

// Low 32 bits of a generic pointer into LDS == wave-relative LDS byte address
// (ISA 10.2: flat->LDS takes addr[31:0]).
DEV unsigned int lds_off_u32(const void* p) {
  return (unsigned int)(unsigned long long)p;
}

// Async global -> LDS copy, 16 bytes per lane, tracked by ASYNCcnt.
DEV void async_g2l_b128(unsigned int lds_byte_off, const void* gptr) {
  asm volatile("global_load_async_to_lds_b128 %0, %1, off"
               :
               : "v"(lds_byte_off), "v"((unsigned long long)gptr)
               : "memory");
}

DEV void wait_asynccnt0() {
  asm volatile("s_wait_asynccnt 0x0" ::: "memory");
}

// Tensor Data Mover: 2D tile (tile_w x tile_h elements of 2 bytes) from a
// strided global tensor into a contiguous row-major LDS tile.
DEV void tdm_load_2d_bf16(unsigned int lds_byte_off, const void* gptr,
                          unsigned int tile_w, unsigned int tile_h,
                          unsigned int row_stride_elems) {
  const unsigned long long ga = (unsigned long long)gptr;
  v4u g0;
  g0[0] = 1u;                                   // count=1, user descriptor
  g0[1] = lds_byte_off;                         // lds_addr
  g0[2] = (unsigned int)(ga & 0xffffffffu);     // global_addr[31:0]
  g0[3] = (unsigned int)((ga >> 32) & 0x01ffffffu) | (2u << 30);  // type=2
  v8i g1;
  g1[0] = (int)(1u << 16);                      // data_size=1 -> 2 bytes
  g1[1] = (int)((tile_w & 0xffffu) << 16);      // tensor_dim0[15:0]
  g1[2] = (int)(((tile_w >> 16) & 0xffffu) |    // tensor_dim0[31:16]
                ((tile_h & 0xffffu) << 16));    // tensor_dim1[15:0]
  g1[3] = (int)(((tile_h >> 16) & 0xffffu) |    // tensor_dim1[31:16]
                ((tile_w & 0xffffu) << 16));    // tile_dim0
  g1[4] = (int)(tile_h & 0xffffu);              // tile_dim1 (tile_dim2=0)
  g1[5] = (int)row_stride_elems;                // tensor_dim0_stride[31:0]
  g1[6] = 0;                                    // stride hi + dim1_stride lo
  g1[7] = 0;
  v4i z4 = {0, 0, 0, 0};
#if __clang_major__ >= 23
  v8i z8 = {0, 0, 0, 0, 0, 0, 0, 0};
  __builtin_amdgcn_tensor_load_to_lds(g0, g1, z4, z4, z8, 0);
#else
  __builtin_amdgcn_tensor_load_to_lds(g0, g1, z4, z4, 0);
#endif
}

// A-matrix fragment (16x32 bf16, row-major tile with leading dim `ld`):
// lane<16: row=lane, K={0..7,16..23}; lane>=16: row=lane-16, K={8..15,24..31}.
DEV v16bf frag_a(const __bf16* t, int ld) {
  const int lane = threadIdx.x & 31;
  const __bf16* p = t + (lane & 15) * ld + ((lane >> 4) << 3);
  v8bf lo = *(const v8bf*)p;
  v8bf hi = *(const v8bf*)(p + 16);
  return cat16(lo, hi);
}

// B-matrix fragment (32x16 bf16) from an n-major tile (row index = N):
// n = lane&15, K = {0..7,16..23} (lane<16) / {8..15,24..31} (lane>=16).
DEV v16bf frag_b(const __bf16* t, int ld) {
  const int lane = threadIdx.x & 31;
  const __bf16* p = t + (lane & 15) * ld + ((lane >> 4) << 3);
  v8bf lo = *(const v8bf*)p;
  v8bf hi = *(const v8bf*)(p + 16);
  return cat16(lo, hi);
}

// ---------------------------------------------------------------------------
// f32 -> bf16 conversion (weights)
// ---------------------------------------------------------------------------
__global__ void cvt_f32_bf16_kernel(const float* __restrict__ src,
                                    __bf16* __restrict__ dst, int n) {
  int i = blockIdx.x * blockDim.x + threadIdx.x;
  if (i < n) dst[i] = (__bf16)src[i];
}

// ---------------------------------------------------------------------------
// LayerNorm over E=1024, one token per 256-thread block, bf16 output.
// ---------------------------------------------------------------------------
__global__ __launch_bounds__(256)
void layernorm_bf16_kernel(const float* __restrict__ x,
                           const float* __restrict__ g,
                           const float* __restrict__ be,
                           __bf16* __restrict__ out) {
  __shared__ float red[256];
  const int tok = blockIdx.x;
  const int tid = threadIdx.x;
  const float* row = x + (size_t)tok * 1024;

  float v[4];
  float s = 0.f;
#pragma unroll
  for (int j = 0; j < 4; ++j) { v[j] = row[tid + 256 * j]; s += v[j]; }
  red[tid] = s;
  __syncthreads();
#pragma unroll
  for (int off = 128; off > 0; off >>= 1) {
    if (tid < off) red[tid] += red[tid + off];
    __syncthreads();
  }
  const float mean = red[0] * (1.f / 1024.f);
  __syncthreads();

  float s2 = 0.f;
#pragma unroll
  for (int j = 0; j < 4; ++j) { float d = v[j] - mean; s2 += d * d; }
  red[tid] = s2;
  __syncthreads();
#pragma unroll
  for (int off = 128; off > 0; off >>= 1) {
    if (tid < off) red[tid] += red[tid + off];
    __syncthreads();
  }
  const float rstd = rsqrtf(red[0] * (1.f / 1024.f) + 1e-5f);

#pragma unroll
  for (int j = 0; j < 4; ++j) {
    const int c = tid + 256 * j;
    out[(size_t)tok * 1024 + c] = (__bf16)((v[j] - mean) * rstd * g[c] + be[c]);
  }
}

// ---------------------------------------------------------------------------
// GEMM:  C[M,N] = A[M,K] * B[N,K]^T   (both bf16 row-major)
// Block: 256 threads = 8 waves (2 x 4), block tile 64 x 128, K-step 32.
// Tiles staged with global_load_async_to_lds_b128 (ASYNCcnt).
// Epilogue: either bf16 store (outb) or f32 store + residual add (outf/res).
// ---------------------------------------------------------------------------
#define GLDT 40  // LDS leading dim (bf16 elems): 80B row stride, 16B aligned

__global__ __launch_bounds__(256)
void gemm_bf16nt_kernel(const __bf16* __restrict__ A,
                        const __bf16* __restrict__ Bw,
                        int M, int N, int K,
                        __bf16* __restrict__ outb,
                        float* __restrict__ outf,
                        const float* __restrict__ res) {
  __shared__ __align__(16) __bf16 At[64 * GLDT];
  __shared__ __align__(16) __bf16 Bt[128 * GLDT];

  const int tid = threadIdx.x;
  const int lane = tid & 31;
  const int w = tid >> 5;
  const int wm = w >> 2, wn = w & 3;
  const int m0 = blockIdx.y * 64;
  const int n0 = blockIdx.x * 128;
  const int lrow = tid >> 2;
  const int lseg = (tid & 3) << 3;

  // per-thread staging addresses (16B per thread per buffer per K-step)
  const unsigned int at_off  = lds_off_u32(At + lrow * GLDT + lseg);
  const unsigned int bt_off0 = lds_off_u32(Bt + lrow * GLDT + lseg);
  const unsigned int bt_off1 = lds_off_u32(Bt + (lrow + 64) * GLDT + lseg);
  const __bf16* ag  = A  + (size_t)(m0 + lrow) * K + lseg;
  const __bf16* bg0 = Bw + (size_t)(n0 + lrow) * K + lseg;
  const __bf16* bg1 = Bw + (size_t)(n0 + lrow + 64) * K + lseg;

  v8f acc[2][2] = {};

  for (int k0 = 0; k0 < K; k0 += 32) {
    __syncthreads();  // previous tile fully consumed
    async_g2l_b128(at_off,  ag  + k0);
    async_g2l_b128(bt_off0, bg0 + k0);
    async_g2l_b128(bt_off1, bg1 + k0);
    wait_asynccnt0();
    __syncthreads();

    v16bf a0 = frag_a(At + (wm * 32 + 0)  * GLDT, GLDT);
    v16bf a1 = frag_a(At + (wm * 32 + 16) * GLDT, GLDT);
    v16bf b0 = frag_b(Bt + (wn * 32 + 0)  * GLDT, GLDT);
    v16bf b1 = frag_b(Bt + (wn * 32 + 16) * GLDT, GLDT);
    acc[0][0] = wmma_bf16(a0, b0, acc[0][0]);
    acc[0][1] = wmma_bf16(a0, b1, acc[0][1]);
    acc[1][0] = wmma_bf16(a1, b0, acc[1][0]);
    acc[1][1] = wmma_bf16(a1, b1, acc[1][1]);
  }

  // Epilogue. C layout: VGPR r, lane -> row = r + 8*(lane/16), col = lane%16.
  const int half = lane >> 4, nn = lane & 15;
#pragma unroll
  for (int mt = 0; mt < 2; ++mt)
#pragma unroll
    for (int nt = 0; nt < 2; ++nt)
#pragma unroll
      for (int r = 0; r < 8; ++r) {
        const size_t row = (size_t)(m0 + wm * 32 + mt * 16 + half * 8 + r);
        const size_t col = (size_t)(n0 + wn * 32 + nt * 16 + nn);
        const float v = acc[mt][nt][r];
        if (outf != nullptr) {
          outf[row * N + col] = v + (res ? res[row * N + col] : 0.f);
        } else {
          outb[row * N + col] = (__bf16)v;
        }
      }
}

// ---------------------------------------------------------------------------
// Flash attention: grid (T/64, H, B), 128 threads = 4 waves x 16 query rows.
// qkv bf16 [token(T*B), 3E]  (q: +0, k: +1024, v: +2048; head offset h*64)
// Q/K tiles fetched by the Tensor Data Mover; V transposed manually.
// ---------------------------------------------------------------------------
__global__ __launch_bounds__(128)
void attn_flash_kernel(const __bf16* __restrict__ qkv,
                       __bf16* __restrict__ ctx) {
  const int Bn = 2;
  const int qt = blockIdx.x;
  const int h  = blockIdx.y;
  const int b  = blockIdx.z;
  const int tid = threadIdx.x, lane = tid & 31, w = tid >> 5;
  const int half = lane >> 4, nn = lane & 15;

  __shared__ __align__(16) __bf16 Qt[64 * 64];      // [q-row][dim]
  __shared__ __align__(16) __bf16 Kt[64 * 64];      // [key][dim]
  __shared__ __align__(16) __bf16 Vt[64 * 64];      // [dim][key] (transposed)
  __shared__ __align__(16) __bf16 Pb[4][16 * 64];   // per-wave P scratch

  const int lr = tid >> 1;                 // 0..63 load row
  const size_t headoff = (size_t)h * 64;
  const unsigned int rowstride = (unsigned int)(Bn * 3072);  // 6144 elems

  // ---- stage Q tile via TDM (64 rows x 64 dims, row stride 6144) ----
  if (w == 0) {
    const __bf16* qg = qkv + ((size_t)(qt * 64) * Bn + b) * 3072 + headoff;
    tdm_load_2d_bf16(lds_off_u32(Qt), qg, 64, 64, rowstride);
    __builtin_amdgcn_s_wait_tensorcnt(0);
  }
  __syncthreads();
  v16bf qa0 = frag_a(Qt + (w * 16) * 64 + 0, 64);
  v16bf qa1 = frag_a(Qt + (w * 16) * 64 + 32, 64);

  float mrow[8], lsum[8];
  v8f acc[4] = {};
#pragma unroll
  for (int r = 0; r < 8; ++r) { mrow[r] = -1e30f; lsum[r] = 0.f; }

  for (int s0 = 0; s0 < 2048; s0 += 64) {
    __syncthreads();
    // ---- K chunk via TDM, V chunk manual transpose into [dim][key] ----
    if (w == 0) {
      const __bf16* kg = qkv + 1024 + ((size_t)s0 * Bn + b) * 3072 + headoff;
      tdm_load_2d_bf16(lds_off_u32(Kt), kg, 64, 64, rowstride);
    }
#pragma unroll
    for (int j = 0; j < 4; ++j) {
      const int d = (((tid & 1) << 2) + j) << 3;
      const size_t base = ((size_t)(s0 + lr) * Bn + b) * 3072 + headoff + d;
      uint4 vv = *(const uint4*)(qkv + 2048 + base);
      const __bf16* vb = (const __bf16*)&vv;
#pragma unroll
      for (int e = 0; e < 8; ++e) Vt[(d + e) * 64 + lr] = vb[e];
    }
    if (w == 0) __builtin_amdgcn_s_wait_tensorcnt(0);
    __syncthreads();

    // ---- S = Q @ K^T ----
    v8f S[4];
#pragma unroll
    for (int nt = 0; nt < 4; ++nt) {
      v8f c = {};
      c = wmma_bf16(qa0, frag_b(Kt + (nt * 16) * 64 + 0, 64), c);
      c = wmma_bf16(qa1, frag_b(Kt + (nt * 16) * 64 + 32, 64), c);
      S[nt] = c;
    }

    // ---- online softmax (row = r + 8*half lives in one 16-lane group) ----
#pragma unroll
    for (int r = 0; r < 8; ++r) {
      float mx = -1e30f;
#pragma unroll
      for (int nt = 0; nt < 4; ++nt) {
        S[nt][r] *= 0.125f;  // HD^-0.5
        mx = fmaxf(mx, S[nt][r]);
      }
      mx = fmaxf(mx, __shfl_xor(mx, 1, 16));
      mx = fmaxf(mx, __shfl_xor(mx, 2, 16));
      mx = fmaxf(mx, __shfl_xor(mx, 4, 16));
      mx = fmaxf(mx, __shfl_xor(mx, 8, 16));
      const float mnew = fmaxf(mrow[r], mx);
      const float alpha = __expf(mrow[r] - mnew);
      float rs = 0.f;
#pragma unroll
      for (int nt = 0; nt < 4; ++nt) {
        const float p = __expf(S[nt][r] - mnew);
        rs += p;
        Pb[w][(r + half * 8) * 64 + nt * 16 + nn] = (__bf16)p;
      }
      rs += __shfl_xor(rs, 1, 16);
      rs += __shfl_xor(rs, 2, 16);
      rs += __shfl_xor(rs, 4, 16);
      rs += __shfl_xor(rs, 8, 16);
      lsum[r] = lsum[r] * alpha + rs;
      mrow[r] = mnew;
#pragma unroll
      for (int nt = 0; nt < 4; ++nt) acc[nt][r] *= alpha;
    }

    // ---- acc += P @ V ----
#pragma unroll
    for (int kc = 0; kc < 2; ++kc) {
      v16bf pa = frag_a(&Pb[w][0] + kc * 32, 64);
#pragma unroll
      for (int nt = 0; nt < 4; ++nt) {
        v16bf vb = frag_b(Vt + (nt * 16) * 64 + kc * 32, 64);
        acc[nt] = wmma_bf16(pa, vb, acc[nt]);
      }
    }
  }

  // ---- normalize + write ctx ----
#pragma unroll
  for (int nt = 0; nt < 4; ++nt)
#pragma unroll
    for (int r = 0; r < 8; ++r) {
      const size_t t = (size_t)(qt * 64 + w * 16 + half * 8 + r);
      const float o = acc[nt][r] / lsum[r];
      ctx[(t * Bn + b) * 1024 + headoff + nt * 16 + nn] = (__bf16)o;
    }
}

// ---------------------------------------------------------------------------
// Launch
// ---------------------------------------------------------------------------
extern "C" void kernel_launch(void* const* d_in, const int* in_sizes, int n_in,
                              void* d_out, int out_size, void* d_ws,
                              size_t ws_size, hipStream_t stream) {
  (void)in_sizes; (void)n_in; (void)out_size; (void)ws_size;

  const float* query = (const float*)d_in[0];
  const float* gamma = (const float*)d_in[1];
  const float* beta  = (const float*)d_in[2];
  const float* w_in  = (const float*)d_in[3];   // [3072,1024]
  const float* w_out = (const float*)d_in[4];   // [1024,1024]
  float* out = (float*)d_out;

  char* ws = (char*)d_ws;
  __bf16* ln_b   = (__bf16*)(ws + 0);          // 4096*1024*2 =  8 MB
  __bf16* win_b  = (__bf16*)(ws + 8388608);    // 3072*1024*2 =  6 MB
  __bf16* wout_b = (__bf16*)(ws + 14680064);   // 1024*1024*2 =  2 MB
  __bf16* qkv_b  = (__bf16*)(ws + 16777216);   // 4096*3072*2 = 24 MB
  __bf16* ctx_b  = (__bf16*)(ws + 41943040);   // 4096*1024*2 =  8 MB

  cvt_f32_bf16_kernel<<<(3072 * 1024) / 256, 256, 0, stream>>>(w_in, win_b,
                                                               3072 * 1024);
  cvt_f32_bf16_kernel<<<(1024 * 1024) / 256, 256, 0, stream>>>(w_out, wout_b,
                                                               1024 * 1024);
  layernorm_bf16_kernel<<<4096, 256, 0, stream>>>(query, gamma, beta, ln_b);

  // qkv = ln @ W_in^T : [4096,3072]
  gemm_bf16nt_kernel<<<dim3(24, 64), 256, 0, stream>>>(
      ln_b, win_b, 4096, 3072, 1024, qkv_b, nullptr, nullptr);

  // ctx = softmax(q k^T / 8) v : [4096,1024]
  attn_flash_kernel<<<dim3(32, 16, 2), 128, 0, stream>>>(qkv_b, ctx_b);

  // out = query + ctx @ W_out^T : [4096,1024] f32
  gemm_bf16nt_kernel<<<dim3(8, 64), 256, 0, stream>>>(
      ctx_b, wout_b, 4096, 1024, 1024, nullptr, out, query);
}
